// DropLearner_8040178778537
// MI455X (gfx1250) — matrix operand order repaired and checked by reading, not compiled, not run on hardware
//
#include <hip/hip_runtime.h>
#include <hip/hip_bf16.h>
#include <stddef.h>

typedef __attribute__((ext_vector_type(16))) __bf16 v16bf;
typedef __attribute__((ext_vector_type(8)))  float  v8f;

#if defined(__has_builtin)
#  if __has_builtin(__builtin_amdgcn_cvt_pk_bf16_f32)
#    define USE_PK_CVT 1
#  endif
#  if __has_builtin(__builtin_amdgcn_sched_group_barrier)
#    define USE_SGB 1
#  endif
#endif

#define NODE_DIM 128
#define XCOLS    384          // 3*NODE_DIM
#define NROWS    192          // 3*MLP_DIM
#define N_REL    32
#define TILE_E   128
#define XPADH    408          // halves per X row (816 B, 16B aligned, bank-staggered)
#define XROWB    (XPADH * 2)  // 816 bytes per X row
#define WPADH    392          // halves per W row (784 B, 16B aligned, bank-staggered)
#define WROWB    (WPADH * 2)  // 784 bytes per W row
#define NTSTEP   (16 * WROWB) // 12544 B between consecutive n-tiles
#define W1_DWORDS (NROWS * XCOLS / 2)   // 36864 packed-bf16 dwords

// W first: all B-fragment DS offsets fit the 16-bit DS immediate.
struct Smem {
    unsigned short W[NROWS][WPADH];    // W1 row-major, bf16                   (150528 B)
    unsigned short X[TILE_E][XPADH];   // gathered [head|tail|rel] rows, bf16  (104448 B)
    int            rows[TILE_E][3];    // gather row indices                   (1536 B)
    float          wpart[2][TILE_E];   // per-n-half partial logits            (1024 B)
};

__device__ __forceinline__ unsigned int pk_bf16(float a, float b) {
#ifdef USE_PK_CVT
    typedef __attribute__((ext_vector_type(2))) __bf16 v2bf;
    union { v2bf v; unsigned int u; } r;
    r.v = __builtin_amdgcn_cvt_pk_bf16_f32(a, b);
    return r.u;
#else
    unsigned int ua = __float_as_uint(a);
    unsigned int ub = __float_as_uint(b);
    ua = (ua + 0x7FFFu + ((ua >> 16) & 1u)) >> 16;   // RNE
    ub = (ub + 0x7FFFu + ((ub >> 16) & 1u)) >> 16;
    return ua | (ub << 16);
#endif
}

// One-time W1 fp32 -> packed bf16 conversion into workspace.
__global__ __launch_bounds__(256) void convert_w1(const float* __restrict__ W1,
                                                  unsigned int* __restrict__ w1bf) {
    int d = blockIdx.x * 256 + threadIdx.x;
    if (d < W1_DWORDS) {
        float2 w = reinterpret_cast<const float2*>(W1)[d];
        w1bf[d] = pk_bf16(w.x, w.y);
    }
}

// One 8-wave block per WGP (257 KB LDS) -> 2 waves/SIMD is the occupancy cap.
__global__ __launch_bounds__(256)
__attribute__((amdgpu_waves_per_eu(2)))
void drop_main(
    const int* __restrict__ edge_index, const int* __restrict__ edge_type,
    const float* __restrict__ all_embed, const float* __restrict__ relation_emb,
    const float* __restrict__ u, const float* __restrict__ W1,
    const unsigned int* __restrict__ w1bf,   // may be null -> convert in-kernel
    const float* __restrict__ b1, const float* __restrict__ W2,
    const float* __restrict__ b2, float* __restrict__ out,
    int E, int n_nodes, int ntiles)
{
    extern __shared__ char smem_raw[];
    Smem* S = reinterpret_cast<Smem*>(smem_raw);

    const int tid = threadIdx.x;

    // ---- Stage W1 -> LDS bf16 (padded rows), once per persistent block ----
    unsigned int* Wd = reinterpret_cast<unsigned int*>(&S->W[0][0]);
    if (w1bf) {
        for (int q = tid; q < NROWS * 48; q += 256) {     // b128 copy path
            int row = q / 48, c = q - row * 48;
            uint4 v = reinterpret_cast<const uint4*>(w1bf)[q];
            *reinterpret_cast<uint4*>(&Wd[row * (WPADH / 2) + c * 4]) = v;
        }
    } else {
        const float2* Wf = reinterpret_cast<const float2*>(W1);
        for (int d = tid; d < W1_DWORDS; d += 256) {      // fallback convert
            float2 w = Wf[d];
            int row = d / (XCOLS / 2);
            int c   = d - row * (XCOLS / 2);
            Wd[row * (WPADH / 2) + c] = pk_bf16(w.x, w.y);
        }
    }

    const int lane = tid & 31;
    const int wv   = tid >> 5;
    const int mg   = wv >> 1;        // m-group 0..3 -> edge rows mg*32..+31
    const int nh   = wv & 1;         // n-half: n-tiles nh*6 .. nh*6+5
    const int lm   = lane & 15;
    const int hi16 = lane >> 4;      // 0: lanes 0-15, 1: lanes 16-31

    // preload per-lane column constants (n = (nh*6+nt)*16 + lm)
    float b1v[6], w2v[6];
    #pragma unroll
    for (int nt = 0; nt < 6; ++nt) {
        int n = (nh * 6 + nt) * 16 + lm;
        b1v[nt] = b1[n];
        w2v[nt] = W2[n];
    }
    const float b2v = b2[0];

    union Frag { uint4 q[2]; v16bf v; };

    // ---- Persistent loop over 128-edge tiles ----
    for (int tile = blockIdx.x; tile < ntiles; tile += gridDim.x) {
        const int e0 = tile * TILE_E;

        // stage gather indices
        if (tid < TILE_E * 3) {
            int r = tid / 3, s = tid - r * 3;
            int e = e0 + r;
            int row = 0;
            if (e < E) {
                if (s == 0)      row = edge_index[e];
                else if (s == 1) row = edge_index[E + e];
                else             row = edge_type[e];
            }
            int lim = (s == 2) ? N_REL : n_nodes;
            if (row < 0) row = 0;
            if (row >= lim) row = lim - 1;
            S->rows[r][s] = row;
        }
        __syncthreads();   // B1: rows ready; previous GEMM complete -> X reusable

        // gather 128 x [head|tail|rel] rows -> LDS bf16 (48 float4-chunks/thread)
        #pragma unroll 4
        for (int i = 0; i < 48; ++i) {
            int c  = tid + i * 256;
            int p  = c >> 5, fq = c & 31;
            int r  = p / 3,  s  = p - r * 3;
            int row = S->rows[r][s];
            const float* base = (s < 2) ? all_embed : relation_emb;
            float4 v = reinterpret_cast<const float4*>(base + (long long)row * NODE_DIM)[fq];
            unsigned int* dst = reinterpret_cast<unsigned int*>(&S->X[r][s * NODE_DIM + fq * 4]);
            dst[0] = pk_bf16(v.x, v.y);
            dst[1] = pk_bf16(v.z, v.w);
        }
        __syncthreads();   // B2: X tile ready

        // ---- Per-wave GEMM: M=32 (2 m-tiles), N=96 (6 n-tiles), K=384 ----
        // Each B fragment feeds two WMMAs -> B LDS traffic halved per edge.
        // Opaque A-offset: forces one indivisible base register so all four
        // A loads use small immediates (0/32/13056/13088) -> no per-k-step
        // address VALU and no WMMA->VALU hazard NOPs.
        unsigned int aoffs = (unsigned int)offsetof(Smem, X)
                           + (unsigned int)((mg * 32 + lm) * XROWB + hi16 * 16);
        asm("" : "+v"(aoffs));
        const char* ab = (const char*)S + aoffs;
        const char* bb = (const char*)S->W + nh * (6 * NTSTEP) + lm * WROWB + hi16 * 16;

        v8f acc0[6] = {}, acc1[6] = {};
        Frag a0, a1, b[3];

        #pragma unroll 1
        for (int kt = 0; kt < 12; ++kt) {
            // A fragments for the two m-tiles (row +16 -> +13056 B)
            a0.q[0] = *reinterpret_cast<const uint4*>(ab);
            a0.q[1] = *reinterpret_cast<const uint4*>(ab + 32);
            a1.q[0] = *reinterpret_cast<const uint4*>(ab + 16 * XROWB);
            a1.q[1] = *reinterpret_cast<const uint4*>(ab + 16 * XROWB + 32);
            // B pipeline prologue: 3 fragments in flight
            b[0].q[0] = *reinterpret_cast<const uint4*>(bb);
            b[0].q[1] = *reinterpret_cast<const uint4*>(bb + 32);
            b[1].q[0] = *reinterpret_cast<const uint4*>(bb + NTSTEP);
            b[1].q[1] = *reinterpret_cast<const uint4*>(bb + NTSTEP + 32);
            b[2].q[0] = *reinterpret_cast<const uint4*>(bb + 2 * NTSTEP);
            b[2].q[1] = *reinterpret_cast<const uint4*>(bb + 2 * NTSTEP + 32);
            #pragma unroll
            for (int nt = 0; nt < 6; ++nt) {
                acc0[nt] = __builtin_amdgcn_wmma_f32_16x16x32_bf16(
                    false, a0.v, false, b[nt % 3].v, (short)0, acc0[nt], false, false);
                acc1[nt] = __builtin_amdgcn_wmma_f32_16x16x32_bf16(
                    false, a1.v, false, b[nt % 3].v, (short)0, acc1[nt], false, false);
                if (nt < 3) {   // refill slot after both consumers issued
                    b[nt].q[0] = *reinterpret_cast<const uint4*>(bb + (nt + 3) * NTSTEP);
                    b[nt].q[1] = *reinterpret_cast<const uint4*>(bb + (nt + 3) * NTSTEP + 32);
                }
            }
            ab += 64;   // 32 halves per k-step
            bb += 64;
#ifdef USE_SGB
            // Pin DS-read / WMMA interleave: 10-load prologue, then
            // (2 WMMA, 2 loads) x3, then the remaining 6 WMMA.
            __builtin_amdgcn_sched_group_barrier(0x100, 10, 0);
            __builtin_amdgcn_sched_group_barrier(0x008, 2, 0);
            __builtin_amdgcn_sched_group_barrier(0x100, 2, 0);
            __builtin_amdgcn_sched_group_barrier(0x008, 2, 0);
            __builtin_amdgcn_sched_group_barrier(0x100, 2, 0);
            __builtin_amdgcn_sched_group_barrier(0x008, 2, 0);
            __builtin_amdgcn_sched_group_barrier(0x100, 2, 0);
            __builtin_amdgcn_sched_group_barrier(0x008, 6, 0);
#endif
        }

        // ---- Epilogue: bias + ReLU + W2 dot, butterfly-reduce across 16 lanes ----
        float s0[8], s1[8];
        #pragma unroll
        for (int g = 0; g < 8; ++g) {
            float sa = 0.f, sb = 0.f;
            #pragma unroll
            for (int nt = 0; nt < 6; ++nt) {
                sa = fmaf(fmaxf(acc0[nt][g] + b1v[nt], 0.f), w2v[nt], sa);
                sb = fmaf(fmaxf(acc1[nt][g] + b1v[nt], 0.f), w2v[nt], sb);
            }
            sa += __shfl_xor(sa, 1, 32);  sb += __shfl_xor(sb, 1, 32);
            sa += __shfl_xor(sa, 2, 32);  sb += __shfl_xor(sb, 2, 32);
            sa += __shfl_xor(sa, 4, 32);  sb += __shfl_xor(sb, 4, 32);
            sa += __shfl_xor(sa, 8, 32);  sb += __shfl_xor(sb, 8, 32);
            s0[g] = sa;  s1[g] = sb;
        }
        if (lm == 0) {
            #pragma unroll
            for (int g = 0; g < 8; ++g) {
                S->wpart[nh][mg * 32 + hi16 * 8 + g]      = s0[g];
                S->wpart[nh][mg * 32 + 16 + hi16 * 8 + g] = s1[g];
            }
        }
        __syncthreads();   // B3: both n-half partials ready

        // ---- Final gate math (exact fp32, matches reference constants) ----
        if (tid < TILE_E) {
            int e = e0 + tid;
            if (e < E) {
                float wsum = S->wpart[0][tid] + S->wpart[1][tid] + b2v;
                float uu   = u[e];
                const float BIAS = 0.0001f;
                float eps  = (2.f * BIAS - 1.f) * uu + (1.f - BIAS);
                float gate = (logf(eps) - log1pf(-eps) + wsum) * 2.0f; // 1/T, T=0.5
                out[e] = 1.f / (1.f + expf(-gate));
            }
        }
    }
}

extern "C" void kernel_launch(void* const* d_in, const int* in_sizes, int n_in,
                              void* d_out, int out_size, void* d_ws, size_t ws_size,
                              hipStream_t stream) {
    const int*   edge_index   = (const int*)d_in[0];
    const int*   edge_type    = (const int*)d_in[1];
    const float* all_embed    = (const float*)d_in[2];
    const float* relation_emb = (const float*)d_in[3];
    const float* u            = (const float*)d_in[4];
    const float* W1           = (const float*)d_in[5];
    const float* b1           = (const float*)d_in[6];
    const float* W2           = (const float*)d_in[7];
    const float* b2           = (const float*)d_in[8];
    float* out = (float*)d_out;

    const int E       = in_sizes[1];
    const int n_nodes = in_sizes[2] / NODE_DIM;
    const int ntiles  = (E + TILE_E - 1) / TILE_E;

    unsigned int* w1bf = nullptr;
    if (ws_size >= (size_t)W1_DWORDS * sizeof(unsigned int)) {
        w1bf = (unsigned int*)d_ws;
        convert_w1<<<(W1_DWORDS + 255) / 256, 256, 0, stream>>>(W1, w1bf);
    }

    int blocks = ntiles < 1024 ? ntiles : 1024;   // persistent blocks, grid-stride tiles
    const size_t smem = sizeof(Smem);             // ~257 KB of the 320 KB/WGP

    drop_main<<<blocks, 256, smem, stream>>>(edge_index, edge_type, all_embed,
                                             relation_emb, u, W1, w1bf, b1, W2, b2,
                                             out, E, n_nodes, ntiles);
}
